// bcnn2d_65867618451703
// MI455X (gfx1250) — matrix operand order repaired and checked by reading, not compiled, not run on hardware
//
#include <hip/hip_runtime.h>

// CDNA5 / gfx1250 implicit-GEMM grouped conv + BN + ReLU pipeline.
// wave32; WMMA f32 = f16 x f16 16x16x32, f32 accumulate.
// Each wave computes a 16(M) x 64(N) strip: 1 A-fragment -> 4 WMMAs per K chunk.
// Input rows cached in LDS once per block -> im2col built from LDS (no 9x
// global re-read); weight tiles streamed via async global->LDS copies.

typedef __attribute__((ext_vector_type(16))) _Float16 v16h;
typedef __attribute__((ext_vector_type(8)))  _Float16 v8h;
typedef __attribute__((ext_vector_type(4)))  _Float16 v4h;
typedef __attribute__((ext_vector_type(8)))  float    v8f;
typedef __attribute__((ext_vector_type(4)))  float    v4f;
typedef __attribute__((ext_vector_type(4)))  int      v4i;

#define NSEEDS 16
#define NSAMPS 32
#define HW     4096   // 64*64
#define WDIM   64

#if defined(__gfx1250__) && __has_builtin(__builtin_amdgcn_global_load_async_to_lds_b128)
#define HAVE_ASYNC_LDS 1
typedef __attribute__((address_space(1))) v4i gv4i;   // global int4
typedef __attribute__((address_space(3))) v4i lv4i;   // LDS int4
#else
#define HAVE_ASYNC_LDS 0
#endif

__device__ __forceinline__ void wait_async_lds() {
#if defined(__gfx1250__)
#if __has_builtin(__builtin_amdgcn_s_wait_asynccnt)
    __builtin_amdgcn_s_wait_asynccnt(0);
#else
    asm volatile("s_wait_asynccnt 0" ::: "memory");
#endif
#endif
}

// Sum across each 16-lane half of the wave (lanes 0-15 and 16-31 separately).
__device__ __forceinline__ float halfwave_reduce_add(float v) {
#if defined(__gfx1250__) && __has_builtin(__builtin_amdgcn_ds_swizzle)
    // ds_swizzle group-of-32 xor mode: offset = xor<<10 | or<<5 | and(0x1f)
    v += __int_as_float(__builtin_amdgcn_ds_swizzle(__float_as_int(v), (8 << 10) | 0x1f));
    v += __int_as_float(__builtin_amdgcn_ds_swizzle(__float_as_int(v), (4 << 10) | 0x1f));
    v += __int_as_float(__builtin_amdgcn_ds_swizzle(__float_as_int(v), (2 << 10) | 0x1f));
    v += __int_as_float(__builtin_amdgcn_ds_swizzle(__float_as_int(v), (1 << 10) | 0x1f));
#else
    for (int off = 8; off >= 1; off >>= 1) v += __shfl_xor(v, off, 32);
#endif
    return v;
}

__device__ __forceinline__ float bn_eps() { return 1e-5f; }

// ---------------------------------------------------------------------------
// Pack x: (S,N,C,H,W) f32  ->  act layout [s][c][n][pixel] f16  (128-bit I/O)
// ---------------------------------------------------------------------------
__global__ void pack_x_kernel(const float* __restrict__ x, _Float16* __restrict__ act) {
    const int total4 = NSEEDS * 3 * NSAMPS * HW / 4;
    int i4 = blockIdx.x * blockDim.x + threadIdx.x;
    if (i4 >= total4) return;
    int i  = i4 * 4;
    int p  = i & (HW - 1);
    int t  = i >> 12;
    int n  = t & (NSAMPS - 1);
    int t2 = t >> 5;
    int c  = t2 % 3;
    int s  = t2 / 3;
    v4f xv = *(const v4f*)&x[(((size_t)(s * NSAMPS + n) * 3 + c) << 12) + p];
    v4h o;
#pragma unroll
    for (int j = 0; j < 4; ++j) o[j] = (_Float16)xv[j];
    *(v4h*)&act[i] = o;
}

__global__ void zero_kernel(float* __restrict__ p, int n) {
    int i = blockIdx.x * blockDim.x + threadIdx.x;
    if (i < n) p[i] = 0.f;
}

// ---------------------------------------------------------------------------
// Pack weights: [s*cout][ktot] f32 -> chunk-major f16 tiles
//   wpk[((s*kchunks + kc)*cout + m)*32 + kk], zero-padded K.
// ---------------------------------------------------------------------------
__global__ void pack_w_kernel(const float* __restrict__ w, _Float16* __restrict__ wpk,
                              int cout, int ktot, int kchunks) {
    int i = blockIdx.x * blockDim.x + threadIdx.x;
    int total = NSEEDS * kchunks * cout * 32;
    if (i >= total) return;
    int kk = i & 31;
    int r  = i >> 5;
    int m  = r % cout;
    int kc = (r / cout) % kchunks;
    int s  = (r / cout) / kchunks;
    int kg = kc * 32 + kk;
    float v = (kg < ktot) ? w[(size_t)(s * cout + m) * ktot + kg] : 0.f;
    wpk[i] = (_Float16)v;
}

// ---------------------------------------------------------------------------
// Conv (implicit GEMM, WMMA) + bias + per-channel sum/sumsq accumulation.
//   act    : [s][cin][n][4096] f16
//   wpk    : chunk-major packed weights (see pack_w_kernel)
//   preact : [s][cout][n][4096] f32
// Block: one (seed, sample, 64-pixel row). Wave w owns channels [16w,16w+16).
// ---------------------------------------------------------------------------
template<int CIN, int COUT>
__global__ __launch_bounds__((COUT / 16) * 32)
void conv_bn_stats_kernel(const _Float16* __restrict__ act,
                          const _Float16* __restrict__ wpk,
                          const float* __restrict__ bias,
                          float* __restrict__ preact,
                          float* __restrict__ gsum,
                          float* __restrict__ gsqs) {
    constexpr int KTOT    = CIN * 9;
    constexpr int KCHUNKS = (KTOT + 31) / 32;
    constexpr int MT      = COUT / 16;
    constexpr int THREADS = MT * 32;

    __shared__ __align__(16) _Float16 Wlds[COUT * 32];        // [m][k] chunk tile
    __shared__ __align__(16) _Float16 Blds[64][32];           // [pixel][k]
    __shared__ __align__(16) _Float16 Rows[CIN * 3 * WDIM];   // input row cache
    __shared__ float sBias[COUT];

    const int t     = threadIdx.x;
    const int lane  = t & 31;
    const int mtile = t >> 5;
    const int s     = blockIdx.z;
    const int n     = blockIdx.y;
    const int y     = blockIdx.x;          // full 64-pixel row

    for (int i = t; i < COUT; i += THREADS) sBias[i] = bias[s * COUT + i];

    // ---- Row cache: rows y-1..y+1 for all CIN channels, 128-bit copies. ----
    {
        constexpr int RUNITS = CIN * 3 * (WDIM / 8);   // v8h units
        const size_t abase = ((size_t)(s * CIN) * NSAMPS + n) << 12;
        for (int u = t; u < RUNITS; u += THREADS) {
            int c   = u / 24;
            int rem = u % 24;
            int dy  = rem >> 3;
            int seg = rem & 7;
            int iy  = y + dy - 1;
            v8h vv = {};
            if ((unsigned)iy < (unsigned)WDIM)
                vv = *(const v8h*)&act[abase + (((size_t)c * NSAMPS) << 12) + iy * WDIM + seg * 8];
            *(v8h*)&Rows[(c * 3 + dy) * WDIM + seg * 8] = vv;
        }
    }

    const _Float16* wsrc = wpk + (size_t)s * KCHUNKS * COUT * 32;

    v8f acc[4] = {};

    for (int kc = 0; kc < KCHUNKS; ++kc) {
        __syncthreads();

        // ---- Stage A: contiguous COUT*32 f16 tile (async copy to LDS). ----
        const _Float16* wt = wsrc + (size_t)kc * COUT * 32;
#if HAVE_ASYNC_LDS
        for (int i = t; i < COUT * 4; i += THREADS) {
            __builtin_amdgcn_global_load_async_to_lds_b128(
                (gv4i*)(wt + i * 8), (lv4i*)(&Wlds[i * 8]), 0, 0);
        }
#else
        for (int i = t; i < COUT * 4; i += THREADS)
            *(v8h*)&Wlds[i * 8] = *(const v8h*)&wt[i * 8];
#endif

        // ---- Stage B: im2col 32 K x 64 px from the LDS row cache. ----
        const int kbase = kc * 32;
        for (int i = t; i < 32 * 64; i += THREADS) {
            int px = i & 63, kk = i >> 6, kg = kbase + kk;
            _Float16 v = (_Float16)0.f;
            if (kg < KTOT) {
                int c  = kg / 9;
                int rr = kg - c * 9;
                int dy = rr / 3;
                int ix = px + (rr - dy * 3) - 1;
                if ((unsigned)ix < (unsigned)WDIM)
                    v = Rows[(c * 3 + dy) * WDIM + ix];
            }
            Blds[px][kk] = v;
        }
#if HAVE_ASYNC_LDS
        wait_async_lds();
#endif
        __syncthreads();

        // Fragments per ISA 16-bit A/B layouts (wave32):
        //  A: lanes 0-15 row m hold K {0..7,16..23}; lanes 16-31 K {8..15,24..31}
        //  B: lanes 0-15 col n hold K {0..7,16..23}; lanes 16-31 K {8..15,24..31}
        const int row  = lane & 15;
        const int asel = (lane < 16) ? 0 : 8;
        const int bsel = (lane < 16) ? 0 : 8;
        const _Float16* wrow = &Wlds[(mtile * 16 + row) * 32];
        const v8h alo = *(const v8h*)&wrow[asel];
        const v8h ahi = *(const v8h*)&wrow[asel + 16];
        v16h a;
#pragma unroll
        for (int j = 0; j < 8; ++j) { a[j] = alo[j]; a[j + 8] = ahi[j]; }

#pragma unroll
        for (int nt = 0; nt < 4; ++nt) {
            const _Float16* brow = &Blds[nt * 16 + row][0];
            const v8h blo = *(const v8h*)&brow[bsel];
            const v8h bhi = *(const v8h*)&brow[bsel + 16];
            v16h b;
#pragma unroll
            for (int j = 0; j < 8; ++j) { b[j] = blo[j]; b[j + 8] = bhi[j]; }
            acc[nt] = __builtin_amdgcn_wmma_f32_16x16x32_f16(
                false, a, false, b, (short)0, acc[nt], false, false);
        }
    }

    // ---- Epilogue: bias, store preact, cross-lane BN partial reduction. ----
#pragma unroll
    for (int r = 0; r < 8; ++r) {
        const int mm = mtile * 16 + r + ((lane < 16) ? 0 : 8);
        const float bsv = sBias[mm];
        const size_t mb = (((size_t)(s * COUT + mm) * NSAMPS + n) << 12) + (size_t)y * WDIM;
        float vsum = 0.f, vsq = 0.f;
#pragma unroll
        for (int nt = 0; nt < 4; ++nt) {
            float v = acc[nt][r] + bsv;
            preact[mb + nt * 16 + (lane & 15)] = v;
            vsum += v;
            vsq  += v * v;
        }
        // Lanes 0-15 (and 16-31) share channel mm: butterfly within each half.
        vsum = halfwave_reduce_add(vsum);
        vsq  = halfwave_reduce_add(vsq);
        if ((lane & 15) == 0) {
            atomicAdd(&gsum[s * COUT + mm], vsum);
            atomicAdd(&gsqs[s * COUT + mm], vsq);
        }
    }
}

// ---------------------------------------------------------------------------
// Fold (sum, sumsq, gamma, beta) -> per-channel (scale, shift).
// ---------------------------------------------------------------------------
__global__ void bn_finalize_kernel(const float* __restrict__ gsum,
                                   const float* __restrict__ gsqs,
                                   const float* __restrict__ gamma,
                                   const float* __restrict__ beta,
                                   float* __restrict__ scale,
                                   float* __restrict__ shift, int nch) {
    int i = blockIdx.x * blockDim.x + threadIdx.x;
    if (i >= nch) return;
    const float inv_cnt = 1.0f / (float)(NSAMPS * HW);
    float mean = gsum[i] * inv_cnt;
    float var  = gsqs[i] * inv_cnt - mean * mean;
    float sc   = gamma[i] * rsqrtf(var + bn_eps());
    scale[i]   = sc;
    shift[i]   = beta[i] - mean * sc;
}

// ---------------------------------------------------------------------------
// Normalize + ReLU -> f16 activations for the next layer (same flat layout).
// ---------------------------------------------------------------------------
template<int COUT>
__global__ void bn_apply_kernel(const float* __restrict__ preact,
                                const float* __restrict__ scale,
                                const float* __restrict__ shift,
                                _Float16* __restrict__ actout) {
    const size_t total4 = (size_t)NSEEDS * COUT * NSAMPS * HW / 4;
    size_t i4 = (size_t)blockIdx.x * blockDim.x + threadIdx.x;
    if (i4 >= total4) return;
    size_t i = i4 * 4;
    int ch = (int)(i >> 17);                 // / (NSAMPS*HW)
    float sc = scale[ch], sh = shift[ch];
    v4f p = *(const v4f*)(preact + i);
    v4h o;
#pragma unroll
    for (int j = 0; j < 4; ++j) o[j] = (_Float16)fmaxf(p[j] * sc + sh, 0.f);
    *(v4h*)(actout + i) = o;
}

// Final layer: normalize + ReLU -> f32 d_out in reference (S,N,C,H,W) layout.
__global__ void bn_apply_final_kernel(const float* __restrict__ preact,
                                      const float* __restrict__ scale,
                                      const float* __restrict__ shift,
                                      float* __restrict__ out) {
    const size_t total4 = (size_t)NSEEDS * 64 * NSAMPS * HW / 4;
    size_t i4 = (size_t)blockIdx.x * blockDim.x + threadIdx.x;
    if (i4 >= total4) return;
    size_t i = i4 * 4;
    int ch   = (int)(i >> 17);               // s*64 + cout
    int p    = (int)(i & (HW - 1));
    int n    = (int)(i >> 12) & (NSAMPS - 1);
    int cout = ch & 63;
    int s    = ch >> 6;
    float sc = scale[ch], sh = shift[ch];
    v4f pr = *(const v4f*)(preact + i);
    v4f o;
#pragma unroll
    for (int j = 0; j < 4; ++j) o[j] = fmaxf(pr[j] * sc + sh, 0.f);
    *(v4f*)&out[((((size_t)s * NSAMPS + n) * 64 + cout) << 12) + p] = o;
}

// ---------------------------------------------------------------------------
extern "C" void kernel_launch(void* const* d_in, const int* in_sizes, int n_in,
                              void* d_out, int out_size, void* d_ws, size_t ws_size,
                              hipStream_t stream) {
    (void)in_sizes; (void)n_in; (void)out_size; (void)ws_size;
    const float* x   = (const float*)d_in[0];
    const float* w1  = (const float*)d_in[1];
    const float* b1  = (const float*)d_in[2];
    const float* g1  = (const float*)d_in[3];
    const float* be1 = (const float*)d_in[4];
    const float* w2  = (const float*)d_in[5];
    const float* b2  = (const float*)d_in[6];
    const float* g2  = (const float*)d_in[7];
    const float* be2 = (const float*)d_in[8];
    const float* w3  = (const float*)d_in[9];
    const float* b3  = (const float*)d_in[10];
    const float* g3  = (const float*)d_in[11];
    const float* be3 = (const float*)d_in[12];
    float* out = (float*)d_out;

    // Workspace layout.
    char* ws = (char*)d_ws;
    const size_t ACT_ELEMS = (size_t)NSEEDS * 64 * NSAMPS * HW;      // max f16 acts
    const size_t ACT_BYTES = ACT_ELEMS * sizeof(_Float16);           // 256 MiB
    _Float16* actA  = (_Float16*)ws;
    _Float16* actB  = (_Float16*)(ws + ACT_BYTES);
    float*    pre   = (float*)(ws + 2 * ACT_BYTES);                  // 512 MiB f32
    float*    stats = (float*)(ws + 2 * ACT_BYTES + ACT_ELEMS * sizeof(float));
    float* gsum  = stats;
    float* gsqs  = stats + 1024;
    float* scale = stats + 2048;
    float* shift = stats + 3072;
    _Float16* wpk = (_Float16*)(stats + 4096);   // <= 1.2 MiB packed weights

    // Pack input into [s][c][n][pixel] f16.
    {
        int total4 = NSEEDS * 3 * NSAMPS * HW / 4;
        pack_x_kernel<<<(total4 + 255) / 256, 256, 0, stream>>>(x, actA);
    }

    const dim3 cgrid(WDIM, NSAMPS, NSEEDS);   // (row, sample, seed)

    // ---- Layer 1: 3 -> 32  (KTOT=27 -> 1 chunk) ----
    {
        int wt = NSEEDS * 1 * 32 * 32;
        pack_w_kernel<<<(wt + 255) / 256, 256, 0, stream>>>(w1, wpk, 32, 27, 1);
        zero_kernel<<<8, 256, 0, stream>>>(gsum, 2048);
        conv_bn_stats_kernel<3, 32><<<cgrid, 2 * 32, 0, stream>>>(actA, wpk, b1, pre, gsum, gsqs);
        bn_finalize_kernel<<<2, 256, 0, stream>>>(gsum, gsqs, g1, be1, scale, shift, NSEEDS * 32);
        size_t total4 = (size_t)NSEEDS * 32 * NSAMPS * HW / 4;
        bn_apply_kernel<32><<<(unsigned)((total4 + 255) / 256), 256, 0, stream>>>(pre, scale, shift, actB);
    }

    // ---- Layer 2: 32 -> 64  (KTOT=288 -> 9 chunks) ----
    {
        int wt = NSEEDS * 9 * 64 * 32;
        pack_w_kernel<<<(wt + 255) / 256, 256, 0, stream>>>(w2, wpk, 64, 288, 9);
        zero_kernel<<<8, 256, 0, stream>>>(gsum, 2048);
        conv_bn_stats_kernel<32, 64><<<cgrid, 4 * 32, 0, stream>>>(actB, wpk, b2, pre, gsum, gsqs);
        bn_finalize_kernel<<<4, 256, 0, stream>>>(gsum, gsqs, g2, be2, scale, shift, NSEEDS * 64);
        size_t total4 = (size_t)NSEEDS * 64 * NSAMPS * HW / 4;
        bn_apply_kernel<64><<<(unsigned)((total4 + 255) / 256), 256, 0, stream>>>(pre, scale, shift, actA);
    }

    // ---- Layer 3: 64 -> 64  (KTOT=576 -> 18 chunks) ----
    {
        int wt = NSEEDS * 18 * 64 * 32;
        pack_w_kernel<<<(wt + 255) / 256, 256, 0, stream>>>(w3, wpk, 64, 576, 18);
        zero_kernel<<<8, 256, 0, stream>>>(gsum, 2048);
        conv_bn_stats_kernel<64, 64><<<cgrid, 4 * 32, 0, stream>>>(actA, wpk, b3, pre, gsum, gsqs);
        bn_finalize_kernel<<<4, 256, 0, stream>>>(gsum, gsqs, g3, be3, scale, shift, NSEEDS * 64);
        size_t total4 = (size_t)NSEEDS * 64 * NSAMPS * HW / 4;
        bn_apply_final_kernel<<<(unsigned)((total4 + 255) / 256), 256, 0, stream>>>(pre, scale, shift, out);
    }
}
